// PointNet2_Class_28123445854328
// MI455X (gfx1250) — matrix operand order repaired and checked by reading, not compiled, not run on hardware
//
#include <hip/hip_runtime.h>
#include <cstdint>
#include <cstddef>

// ---------------------------------------------------------------------------
// PointNet++ classification forward, fused for gfx1250 (MI455X, wave32, WMMA)
// All dense math runs through v_wmma_f32_16x16x32_f16 with f32 accumulation.
// BN (eval mode) is folded into per-channel scale/shift in the GEMM epilogue.
// ---------------------------------------------------------------------------

typedef __attribute__((ext_vector_type(16))) _Float16 v16h;
typedef __attribute__((ext_vector_type(2)))  _Float16 v2h;
typedef __attribute__((ext_vector_type(8)))  float    v8f;

// ---- WMMA fragment loaders (ISA 7.12.2 layouts, wave32) -------------------

// A-matrix 16x32 f16, row-major source with leading dim `ld` (elements).
// base points at element (row 0, k 0) of the tile.
__device__ __forceinline__ v16h load_A_tile(const _Float16* base, int ld, int lane) {
  int row = lane & 15;
  int hi  = (lane >> 4) & 1;         // lanes 16..31
  v16h a;
#pragma unroll
  for (int r = 0; r < 8; ++r) {
    int kb = ((r & 4) ? 16 : 0) + (hi ? 8 : 0) + (r & 3) * 2;  // per ISA A layout
    v2h p = *(const v2h*)(base + row * ld + kb);
    a[2 * r]     = p[0];
    a[2 * r + 1] = p[1];
  }
  return a;
}

// B-matrix 32x16 f16 taken from transposed weights Wt[out][kp] so the two
// K values per VGPR are contiguous in memory (single 4B load per fragment reg).
// wt points at element (k 0, n 0): element (k,n) lives at wt + n*kp + k.
__device__ __forceinline__ v16h load_B_tileT(const _Float16* wt, int kp, int lane) {
  int n  = lane & 15;
  int hi = (lane >> 4) & 1;
  v16h b;
#pragma unroll
  for (int r = 0; r < 8; ++r) {
    int k = ((r & 4) ? 16 : 0) + (hi ? 8 : 0) + (r & 3) * 2;   // same K order as A layout
    v2h p = *(const v2h*)(wt + (size_t)n * kp + k);
    b[2 * r]     = p[0];
    b[2 * r + 1] = p[1];
  }
  return b;
}

__device__ __forceinline__ void atomicMaxFloat(float* addr, float val) {
  int* ia = (int*)addr;
  int cur = __float_as_int(*addr);
  while (__int_as_float(cur) < val) {
    int assumed = cur;
    cur = atomicCAS(ia, assumed, __float_as_int(val));
    if (cur == assumed) break;
  }
}

// ---- Workgroup GEMM: (rows x kp) * Wt(kp x ncols) -> LDS f16 out ----------
// rows = nrt*16, inputs/outputs f16 in LDS, weights f16 in global (transposed).
// Epilogue: v = relu(acc + bias) * scale + shift   (scale=1,shift=0 => plain).
__device__ __forceinline__ void wg_gemm(const _Float16* inBuf, int lda,
                                        _Float16* outBuf, int ldo,
                                        const _Float16* Wt, const float* bias,
                                        const float* scale, const float* shift,
                                        int kp, int ncols, int nrt, bool relu) {
  int wid  = threadIdx.x >> 5;
  int lane = threadIdx.x & 31;
  int nwv  = blockDim.x >> 5;
  int nct  = ncols >> 4;
  int nt   = nct * nrt;
  for (int t = wid; t < nt; t += nwv) {
    int rt = t % nrt, ct = t / nrt;
    v8f acc = {};
    for (int kt = 0; kt < kp; kt += 32) {
      v16h a = load_A_tile(inBuf + rt * 16 * lda + kt, lda, lane);
      v16h b = load_B_tileT(Wt + (size_t)(ct * 16) * kp + kt, kp, lane);
      acc = __builtin_amdgcn_wmma_f32_16x16x32_f16(false, a, false, b,
                                                   (short)0, acc, false, false);
    }
    int n   = lane & 15;
    int col = ct * 16 + n;
    int hi  = (lane >> 4) & 1;
    float bv = bias[col], sc = scale[col], sh = shift[col];
#pragma unroll
    for (int r = 0; r < 8; ++r) {
      float v = acc[r] + bv;
      if (relu) v = v > 0.f ? v : 0.f;
      v = v * sc + sh;
      outBuf[(size_t)(rt * 16 + r + hi * 8) * ldo + col] = (_Float16)v;
    }
  }
}

// Same GEMM but the epilogue folds a row-max and atomically max-reduces into
// pooled[col] (global max pool over points, all rows of this WG share a batch).
__device__ __forceinline__ void wg_gemm_poolmax(const _Float16* inBuf, int lda,
                                                const _Float16* Wt, const float* bias,
                                                const float* scale, const float* shift,
                                                int kp, int ncols, int nrt,
                                                float* pooled) {
  int wid  = threadIdx.x >> 5;
  int lane = threadIdx.x & 31;
  int nwv  = blockDim.x >> 5;
  int nct  = ncols >> 4;
  int nt   = nct * nrt;
  for (int t = wid; t < nt; t += nwv) {
    int rt = t % nrt, ct = t / nrt;
    v8f acc = {};
    for (int kt = 0; kt < kp; kt += 32) {
      v16h a = load_A_tile(inBuf + rt * 16 * lda + kt, lda, lane);
      v16h b = load_B_tileT(Wt + (size_t)(ct * 16) * kp + kt, kp, lane);
      acc = __builtin_amdgcn_wmma_f32_16x16x32_f16(false, a, false, b,
                                                   (short)0, acc, false, false);
    }
    int n   = lane & 15;
    int col = ct * 16 + n;
    float bv = bias[col], sc = scale[col], sh = shift[col];
    float lmax = -1e30f;
#pragma unroll
    for (int r = 0; r < 8; ++r) {
      float v = acc[r] + bv;
      v = v > 0.f ? v : 0.f;
      v = v * sc + sh;
      lmax = fmaxf(lmax, v);
    }
    atomicMaxFloat(&pooled[col], lmax);
  }
}

// ---- Param prep: pad + transpose weights to f16, fold BN ------------------
__global__ void prep_layer(const float* W, const float* bias_in, const float* gamma,
                           const float* beta, const float* mu, const float* var,
                           _Float16* Wt, float* bias, float* scale, float* shift,
                           int in_dim, int in_pad, int out_dim, int out_pad, int hasBN) {
  int i   = blockIdx.x * blockDim.x + threadIdx.x;
  int tot = in_pad * out_pad;
  if (i < tot) {
    int n = i / in_pad, k = i - n * in_pad;
    float v = (k < in_dim && n < out_dim) ? W[(size_t)k * out_dim + n] : 0.f;
    Wt[(size_t)n * in_pad + k] = (_Float16)v;
  }
  if (i < out_pad) {
    if (i < out_dim) {
      bias[i] = bias_in[i];
      if (hasBN) {
        float sc = gamma[i] * rsqrtf(var[i] + 1e-5f);
        scale[i] = sc;
        shift[i] = beta[i] - mu[i] * sc;
      } else {
        scale[i] = 1.f;
        shift[i] = 0.f;
      }
    } else {
      bias[i] = 0.f; scale[i] = 0.f; shift[i] = 0.f;   // padded channels -> 0
    }
  }
}

// ---- Farthest point sampling: one workgroup per cloud ---------------------
__global__ void __launch_bounds__(256) fps_kernel(const float* pos, float* cen,
                                                  int N, int M) {
  int b = blockIdx.x;
  const float* P = pos + (size_t)b * N * 3;
  float* C = cen + (size_t)b * M * 3;
  __shared__ float mind[2048];
  __shared__ float rval[256];
  __shared__ int   ridx[256];
  __shared__ int   sLast;
  int tid = threadIdx.x;
  for (int i = tid; i < N; i += 256) mind[i] = 1e30f;
  if (tid == 0) sLast = 0;
  __syncthreads();
  for (int it = 0; it < M; ++it) {
    int last = sLast;
    float lx = P[last * 3], ly = P[last * 3 + 1], lz = P[last * 3 + 2];
    if (tid == 0) { C[it * 3] = lx; C[it * 3 + 1] = ly; C[it * 3 + 2] = lz; }
    float best = -1e30f; int bi = 0;
    for (int i = tid; i < N; i += 256) {
      float dx = P[i * 3] - lx, dy = P[i * 3 + 1] - ly, dz = P[i * 3 + 2] - lz;
      float d  = dx * dx + dy * dy + dz * dz;
      float mv = fminf(mind[i], d);
      mind[i] = mv;
      if (mv > best) { best = mv; bi = i; }
    }
    rval[tid] = best; ridx[tid] = bi;
    __syncthreads();
    for (int s = 128; s > 0; s >>= 1) {
      if (tid < s && rval[tid + s] > rval[tid]) { rval[tid] = rval[tid + s]; ridx[tid] = ridx[tid + s]; }
      __syncthreads();
    }
    if (tid == 0) sLast = ridx[0];
    __syncthreads();
  }
}

// ---- Ball query: first K neighbors within radius (PointNet++ semantics) ---
__global__ void ball_kernel(const float* pos_src, const float* cen, int* nbr,
                            int Nsrc, int M, int Kq, float r2, int tot) {
  int g = blockIdx.x * blockDim.x + threadIdx.x;
  if (g >= tot) return;
  int b = g / M;
  const float* pos = pos_src + (size_t)b * Nsrc * 3;
  const float* cc  = cen + (size_t)g * 3;
  float cx = cc[0], cy = cc[1], cz = cc[2];
  int* out = nbr + (size_t)g * Kq;
  int cnt = 0;
  for (int n = 0; n < Nsrc && cnt < Kq; ++n) {
    float dx = pos[n * 3] - cx, dy = pos[n * 3 + 1] - cy, dz = pos[n * 3 + 2] - cz;
    float d2 = dx * dx + dy * dy + dz * dz;
    if (d2 <= r2) out[cnt++] = n;
  }
  int pad = cnt > 0 ? out[0] : 0;   // duplicate rows are max-pool invariant
  for (int k = cnt; k < Kq; ++k) out[k] = pad;
}

// ---- Fused SA module MLP: one workgroup (4 waves) per center --------------
// Gathers 64 edge features into LDS, runs 3 WMMA GEMM layers ping-ponging two
// 64x256 f16 LDS buffers, then max-pools over the 64 edges.
#define SA_LD 256
__global__ void __launch_bounds__(128) sa_mlp_kernel(
    const float* x_f32, const _Float16* x_f16, int fx, int Nsrc,
    const float* pos_src, const float* cen, const int* nbr, int M,
    const _Float16* Wt0, const float* b0, const float* s0, const float* h0, int ip0, int o0,
    const _Float16* Wt1, const float* b1, const float* s1, const float* h1, int o1,
    const _Float16* Wt2, const float* b2, const float* s2, const float* h2, int o2,
    _Float16* x_out) {
  __shared__ _Float16 bufA[64 * SA_LD];
  __shared__ _Float16 bufB[64 * SA_LD];
  int bm = blockIdx.x;
  int b = bm / M, m = bm - b * M;
  const float* pos = pos_src + (size_t)b * Nsrc * 3;
  const float* cc  = cen + (size_t)b * M * 3 + (size_t)m * 3;
  const int*   nb  = nbr + (size_t)bm * 64;
  const float*    xf = x_f32 ? x_f32 + (size_t)b * Nsrc * fx : nullptr;
  const _Float16* xh = x_f16 ? x_f16 + (size_t)b * Nsrc * fx : nullptr;
  float cx = cc[0], cy = cc[1], cz = cc[2];
  int in0 = fx + 3;
  int tid = threadIdx.x;

  // Gather edge features [x[nbr] | pos[nbr]-cen], zero-padded to ip0.
  for (int e = tid; e < 64 * ip0; e += blockDim.x) {
    int k = e / ip0, f = e - k * ip0;
    float v = 0.f;
    if (f < in0) {
      int n = nb[k];
      if (f < fx) {
        v = xf ? xf[(size_t)n * fx + f] : (float)xh[(size_t)n * fx + f];
      } else {
        int d = f - fx;
        float cv = (d == 0) ? cx : ((d == 1) ? cy : cz);
        v = pos[(size_t)n * 3 + d] - cv;
      }
    }
    bufA[(size_t)k * SA_LD + f] = (_Float16)v;
  }
  __syncthreads();
  wg_gemm(bufA, SA_LD, bufB, SA_LD, Wt0, b0, s0, h0, ip0, o0, 4, true);
  __syncthreads();
  wg_gemm(bufB, SA_LD, bufA, SA_LD, Wt1, b1, s1, h1, o0, o1, 4, true);
  __syncthreads();
  wg_gemm(bufA, SA_LD, bufB, SA_LD, Wt2, b2, s2, h2, o1, o2, 4, true);
  __syncthreads();
  // Max pool over the 64 edges.
  for (int c = tid; c < o2; c += blockDim.x) {
    _Float16 mx = bufB[c];
    for (int k = 1; k < 64; ++k) {
      _Float16 v = bufB[(size_t)k * SA_LD + c];
      if (v > mx) mx = v;
    }
    x_out[(size_t)bm * o2 + c] = mx;
  }
}

// ---- SA3 pointwise MLP (259->256->512->1024) + fused global max pool ------
__global__ void __launch_bounds__(128) sa3_pool_kernel(
    const _Float16* x2, const float* p2,
    const _Float16* Wt0, const float* b0, const float* s0, const float* h0,
    const _Float16* Wt1, const float* b1, const float* s1, const float* h1,
    const _Float16* Wt2, const float* b2, const float* s2, const float* h2,
    float* pooled) {
  __shared__ _Float16 smem[32 * 512 + 32 * 256];
  _Float16* bufR = smem;              // 32x512 region; also hosts 32x288 gather
  _Float16* buf0 = smem + 32 * 512;   // 32x256
  const int IP = 288;
  int row0 = blockIdx.x * 32;
  int b = row0 >> 8;                  // 256 rows per batch
  int tid = threadIdx.x;
  for (int e = tid; e < 32 * IP; e += blockDim.x) {
    int rr = e / IP, f = e - rr * IP;
    int rg = row0 + rr;
    float v;
    if (f < 256)      v = (float)x2[(size_t)rg * 256 + f];
    else if (f < 259) v = p2[(size_t)rg * 3 + (f - 256)];
    else              v = 0.f;
    bufR[(size_t)rr * IP + f] = (_Float16)v;
  }
  __syncthreads();
  wg_gemm(bufR, IP, buf0, 256, Wt0, b0, s0, h0, 288, 256, 2, true);
  __syncthreads();
  wg_gemm(buf0, 256, bufR, 512, Wt1, b1, s1, h1, 256, 512, 2, true);
  __syncthreads();
  wg_gemm_poolmax(bufR, 512, Wt2, b2, s2, h2, 512, 1024, 2,
                  pooled + (size_t)b * 1024);
}

__global__ void init_pooled(float* p, int n) {
  int i = blockIdx.x * blockDim.x + threadIdx.x;
  if (i < n) p[i] = -1e30f;
}

// ---- Head MLP: 1024->512 relu, 512->256 relu, 256->7 ----------------------
__global__ void __launch_bounds__(128) head_kernel(
    const float* pooled,
    const _Float16* Wt0, const float* b0, const float* s0, const float* h0,
    const _Float16* Wt1, const float* b1, const float* s1, const float* h1,
    const _Float16* Wt2, const float* b2, const float* s2, const float* h2,
    float* out) {
  __shared__ _Float16 bin[16 * 1024];
  __shared__ _Float16 bh1[16 * 512];
  __shared__ _Float16 bh2[16 * 256];
  int tid = threadIdx.x;
  for (int e = tid; e < 16 * 1024; e += blockDim.x) bin[e] = (_Float16)pooled[e];
  __syncthreads();
  wg_gemm(bin, 1024, bh1, 512, Wt0, b0, s0, h0, 1024, 512, 1, true);
  __syncthreads();
  wg_gemm(bh1, 512, bh2, 256, Wt1, b1, s1, h1, 512, 256, 1, true);
  __syncthreads();
  (void)s2; (void)h2;
  if ((threadIdx.x >> 5) == 0) {     // single 16x16 output tile, no relu
    int lane = threadIdx.x & 31;
    v8f acc = {};
    for (int kt = 0; kt < 256; kt += 32) {
      v16h a = load_A_tile(bh2 + kt, 256, lane);
      v16h bb = load_B_tileT(Wt2 + kt, 256, lane);
      acc = __builtin_amdgcn_wmma_f32_16x16x32_f16(false, a, false, bb,
                                                   (short)0, acc, false, false);
    }
    int n  = lane & 15;
    int hi = (lane >> 4) & 1;
    if (n < 7) {
      float bv = b2[n];
#pragma unroll
      for (int r = 0; r < 8; ++r) {
        int mrow = r + hi * 8;       // batch index (B==16 == tile rows)
        out[mrow * 7 + n] = acc[r] + bv;
      }
    }
  }
}

// ---------------------------------------------------------------------------
extern "C" void kernel_launch(void* const* d_in, const int* in_sizes, int n_in,
                              void* d_out, int out_size, void* d_ws, size_t ws_size,
                              hipStream_t stream) {
  (void)in_sizes; (void)n_in; (void)out_size; (void)ws_size;
  const float* x   = (const float*)d_in[0];
  const float* pos = (const float*)d_in[1];
  // params flattened in setup_inputs insertion order:
  // sa1[0..2], sa2[0..2], sa3[0..2], head[0..2]; each layer: W,b,gamma,beta,mu,var
  auto P = [&](int layer, int leaf) -> const float* {
    return (const float*)d_in[2 + layer * 6 + leaf];
  };

  static const int IN[12] = {6, 64, 64, 131, 128, 128, 259, 256, 512, 1024, 512, 256};
  static const int IP[12] = {32, 64, 64, 160, 128, 128, 288, 256, 512, 1024, 512, 256};
  static const int ON[12] = {64, 64, 128, 128, 128, 256, 256, 512, 1024, 512, 256, 7};
  static const int OP[12] = {64, 64, 128, 128, 128, 256, 256, 512, 1024, 512, 256, 16};

  char* cur = (char*)d_ws;
  auto alloc = [&](size_t bytes) -> char* {
    char* p = cur;
    cur += (bytes + 255) & ~(size_t)255;
    return p;
  };

  _Float16* Wt[12];
  float *Bi[12], *Sc[12], *Sh[12];
  for (int l = 0; l < 12; ++l) {
    Wt[l] = (_Float16*)alloc((size_t)IP[l] * OP[l] * sizeof(_Float16));
    Bi[l] = (float*)alloc((size_t)OP[l] * sizeof(float));
    Sc[l] = (float*)alloc((size_t)OP[l] * sizeof(float));
    Sh[l] = (float*)alloc((size_t)OP[l] * sizeof(float));
  }
  const int B = 16, N = 2048, M1 = 1024, M2 = 256, K = 64;
  float*    p1     = (float*)alloc((size_t)B * M1 * 3 * sizeof(float));
  _Float16* x1     = (_Float16*)alloc((size_t)B * M1 * 128 * sizeof(_Float16));
  float*    p2     = (float*)alloc((size_t)B * M2 * 3 * sizeof(float));
  _Float16* x2     = (_Float16*)alloc((size_t)B * M2 * 256 * sizeof(_Float16));
  int*      nbr1   = (int*)alloc((size_t)B * M1 * K * sizeof(int));
  int*      nbr2   = (int*)alloc((size_t)B * M2 * K * sizeof(int));
  float*    pooled = (float*)alloc((size_t)B * 1024 * sizeof(float));

  for (int l = 0; l < 12; ++l) {
    int hasBN = (l < 9) ? 1 : 0;  // head MLP ignores its BN params
    int tot = IP[l] * OP[l];
    prep_layer<<<(tot + 255) / 256, 256, 0, stream>>>(
        P(l, 0), P(l, 1), P(l, 2), P(l, 3), P(l, 4), P(l, 5),
        Wt[l], Bi[l], Sc[l], Sh[l], IN[l], IP[l], ON[l], OP[l], hasBN);
  }

  // SA1
  fps_kernel<<<B, 256, 0, stream>>>(pos, p1, N, M1);
  {
    int tot = B * M1;
    ball_kernel<<<(tot + 127) / 128, 128, 0, stream>>>(pos, p1, nbr1, N, M1, K, 0.04f, tot);
  }
  sa_mlp_kernel<<<B * M1, 128, 0, stream>>>(
      x, (const _Float16*)nullptr, 3, N, pos, p1, nbr1, M1,
      Wt[0], Bi[0], Sc[0], Sh[0], IP[0], ON[0],
      Wt[1], Bi[1], Sc[1], Sh[1], ON[1],
      Wt[2], Bi[2], Sc[2], Sh[2], ON[2],
      x1);

  // SA2
  fps_kernel<<<B, 256, 0, stream>>>(p1, p2, M1, M2);
  {
    int tot = B * M2;
    ball_kernel<<<(tot + 127) / 128, 128, 0, stream>>>(p1, p2, nbr2, M1, M2, K, 0.16f, tot);
  }
  sa_mlp_kernel<<<B * M2, 128, 0, stream>>>(
      (const float*)nullptr, x1, 128, M1, p1, p2, nbr2, M2,
      Wt[3], Bi[3], Sc[3], Sh[3], IP[3], ON[3],
      Wt[4], Bi[4], Sc[4], Sh[4], ON[4],
      Wt[5], Bi[5], Sc[5], Sh[5], ON[5],
      x2);

  // SA3 + global max pool
  init_pooled<<<(B * 1024 + 255) / 256, 256, 0, stream>>>(pooled, B * 1024);
  sa3_pool_kernel<<<(B * M2) / 32, 128, 0, stream>>>(
      x2, p2,
      Wt[6], Bi[6], Sc[6], Sh[6],
      Wt[7], Bi[7], Sc[7], Sh[7],
      Wt[8], Bi[8], Sc[8], Sh[8],
      pooled);

  // Head
  head_kernel<<<1, 128, 0, stream>>>(
      pooled,
      Wt[9],  Bi[9],  Sc[9],  Sh[9],
      Wt[10], Bi[10], Sc[10], Sh[10],
      Wt[11], Bi[11], Sc[11], Sh[11],
      (float*)d_out);
}
